// TopKPoolingLayer_36962488549913
// MI455X (gfx1250) — matrix operand order repaired and checked by reading, not compiled, not run on hardware
//
#include <hip/hip_runtime.h>
#include <stdint.h>

typedef __attribute__((ext_vector_type(16))) _Float16 v16h;
typedef __attribute__((ext_vector_type(8)))  float    v8f;
typedef __attribute__((ext_vector_type(4)))  float    f32x4;

struct SelState {
  float    inv_norm;
  uint32_t prefix;   // radix-select prefix / final threshold key
  uint32_t kneed;    // remaining rank within candidate set -> #ties to take
  uint32_t totalG;   // #elements strictly greater than threshold
};

__device__ __forceinline__ uint32_t f2key(float f) {
  uint32_t u = __float_as_uint(f);
  return u ^ ((u >> 31) ? 0xFFFFFFFFu : 0x80000000u); // monotonic: bigger float -> bigger key
}

// ---------------- init: ||w||, f16 weights, zero hist, select state ----------------
__global__ __launch_bounds__(256) void k_init(const float* __restrict__ w, int C,
                                              SelState* st, _Float16* hw,
                                              uint32_t* hist, uint32_t Ksel) {
  __shared__ float red[256];
  int t = threadIdx.x;
  float v = (t < C) ? w[t] : 0.f;
  if (t < C) hw[t] = (_Float16)v;
  red[t] = v * v;
  __syncthreads();
  for (int s = 128; s > 0; s >>= 1) { if (t < s) red[t] += red[t + s]; __syncthreads(); }
  if (t == 0) {
    st->inv_norm = rsqrtf(red[0]);
    st->prefix = 0u; st->kneed = Ksel; st->totalG = 0u;
  }
  hist[t] = 0u;
}

// ---------------- scores via WMMA f16 (16 rows per wave, K=256 in 8 steps) ----------------
__global__ __launch_bounds__(256) void k_score(const float* __restrict__ x,
                                               const _Float16* __restrict__ hw,
                                               const SelState* __restrict__ st,
                                               float* __restrict__ scores, int N) {
  const int tid  = threadIdx.x;
  const int lane = tid & 31;
  const int wave = tid >> 5;
  const int half = lane >> 4;   // 0: K-low half, 1: K-high half (A layout)
  const int l    = lane & 15;
  const int m0   = (blockIdx.x * 8 + wave) * 16;

  int rowA = m0 + l; if (rowA > N - 1) rowA = N - 1;   // clamp; keep EXEC all-ones
  const float* rp = x + (size_t)rowA * 256;
  const v16h* wv = (const v16h*)hw;

  v8f c = {};
  #pragma unroll
  for (int kb = 0; kb < 256; kb += 32) {
    const int kA0 = kb + (half << 4);        // halves 0..7  : K = kA0..kA0+7
    const int kA1 = kb + (half << 4) + 8;    // halves 8..15 : K = kA1..kA1+7
    f32x4 a0 = *(const f32x4*)(rp + kA0);
    f32x4 a1 = *(const f32x4*)(rp + kA0 + 4);
    f32x4 a2 = *(const f32x4*)(rp + kA1);
    f32x4 a3 = *(const f32x4*)(rp + kA1 + 4);
    v16h a;
    #pragma unroll
    for (int j = 0; j < 4; ++j) {
      a[j]      = (_Float16)a0[j];
      a[4 + j]  = (_Float16)a1[j];
      a[8 + j]  = (_Float16)a2[j];
      a[12 + j] = (_Float16)a3[j];
    }
    // B 32x16: lane<16 holds K=kb..kb+15, lane>=16 holds K=kb+16..kb+31; all columns = w chunk
    v16h b = wv[(kb >> 4) + half];
    c = __builtin_amdgcn_wmma_f32_16x16x32_f16(false, a, false, b, (short)0, c, false, false);
  }

  // D layout: lane(half=0) VGPR r -> row m0+r ; lane(half=1) VGPR r -> row m0+8+r.
  float v = 0.f;
  #pragma unroll
  for (int r = 0; r < 8; ++r) if ((l & 7) == r) v = c[r];
  int writeRow = m0 + (half << 3) + (l & 7);
  if (l < 8 && writeRow < N)
    scores[writeRow] = tanhf(v * st->inv_norm);
}

// ---------------- radix-select round: 256-bin histogram of current byte ----------------
__global__ __launch_bounds__(256) void k_hist(const float* __restrict__ scores, int N,
                                              uint32_t* hist, const SelState* st, int shift) {
  __shared__ uint32_t lh[256];
  int t = threadIdx.x;
  lh[t] = 0u;
  __syncthreads();
  uint32_t maskHi = (shift >= 24) ? 0u : (0xFFFFFFFFu << (shift + 8));
  uint32_t pref = st->prefix & maskHi;
  int i = blockIdx.x * 256 + t;
  if (i < N) {
    uint32_t key = f2key(scores[i]);
    if ((key & maskHi) == pref) atomicAdd(&lh[(key >> shift) & 255u], 1u);
  }
  __syncthreads();
  if (lh[t]) atomicAdd(&hist[t], lh[t]);
}

__global__ __launch_bounds__(256) void k_select(uint32_t* hist, SelState* st, int shift) {
  __shared__ uint32_t lh[256];
  int d = threadIdx.x;
  lh[d] = hist[d];
  hist[d] = 0u;                     // reset for next round
  __syncthreads();
  uint32_t suffix = 0u;             // count of candidates with digit > d
  for (int e = d + 1; e < 256; ++e) suffix += lh[e];
  uint32_t kneed = st->kneed;
  if (suffix < kneed && suffix + lh[d] >= kneed) {   // exactly one thread matches
    st->prefix |= ((uint32_t)d) << shift;
    st->kneed = kneed - suffix;
  }
}

// ---------------- deterministic compaction: count -> scan -> scatter ----------------
__global__ __launch_bounds__(256) void k_count(const float* __restrict__ scores, int N,
                                               const SelState* st,
                                               uint32_t* cntG, uint32_t* cntE) {
  __shared__ uint32_t cg, ce;
  int t = threadIdx.x;
  if (t == 0) { cg = 0u; ce = 0u; }
  __syncthreads();
  int i = blockIdx.x * 256 + t;
  if (i < N) {
    uint32_t key = f2key(scores[i]);
    uint32_t T = st->prefix;
    if (key > T)       atomicAdd(&cg, 1u);
    else if (key == T) atomicAdd(&ce, 1u);
  }
  __syncthreads();
  if (t == 0) { cntG[blockIdx.x] = cg; cntE[blockIdx.x] = ce; }
}

__global__ void k_scan(const uint32_t* cntG, const uint32_t* cntE,
                       uint32_t* baseG, uint32_t* baseE, int nb, SelState* st) {
  if (threadIdx.x == 0 && blockIdx.x == 0) {
    uint32_t g = 0u, e = 0u;
    for (int b = 0; b < nb; ++b) {
      baseG[b] = g; g += cntG[b];
      baseE[b] = e; e += cntE[b];
    }
    st->totalG = g;
  }
}

__global__ __launch_bounds__(256) void k_scatter(const float* __restrict__ scores, int N,
                                                 const SelState* st,
                                                 const uint32_t* baseG, const uint32_t* baseE,
                                                 int* __restrict__ perm, int* __restrict__ mapping) {
  __shared__ uint32_t sc[256];
  int t = threadIdx.x;
  int i = blockIdx.x * 256 + t;
  uint32_t T = st->prefix, rem = st->kneed, totalG = st->totalG;
  bool inb = i < N;
  uint32_t key = inb ? f2key(scores[i]) : 0u;
  bool isG = inb && (key > T);
  bool isT = inb && (key == T);
  uint32_t val = (isG ? 0x10000u : 0u) | (isT ? 1u : 0u);
  sc[t] = val;
  __syncthreads();
  for (int off = 1; off < 256; off <<= 1) {          // Hillis-Steele inclusive scan
    uint32_t add = (t >= off) ? sc[t - off] : 0u;
    __syncthreads();
    sc[t] += add;
    __syncthreads();
  }
  uint32_t excl = sc[t] - val;
  if (inb) {
    int idx = -1;
    if (isG) {
      idx = (int)(baseG[blockIdx.x] + (excl >> 16));
    } else if (isT) {
      uint32_t tr = baseE[blockIdx.x] + (excl & 0xFFFFu);
      if (tr < rem) idx = (int)(totalG + tr);        // first `rem` ties by index
    }
    if (idx >= 0) { perm[idx] = i; mapping[i] = idx; }
    else          { mapping[i] = -1; }
  }
}

// ---------------- gather + gate features ----------------
__global__ __launch_bounds__(64) void k_gather(const float* __restrict__ x,
                                               const int* __restrict__ perm,
                                               const float* __restrict__ scores,
                                               float* __restrict__ out) {
  int r = blockIdx.x;
  int node = perm[r];
  float s = scores[node];
  const f32x4* src = (const f32x4*)(x + (size_t)node * 256);
  f32x4* dst = (f32x4*)(out + (size_t)r * 256);
  dst[threadIdx.x] = src[threadIdx.x] * s;
}

// ---------------- edge filter / remap ----------------
__global__ __launch_bounds__(256) void k_edges(const int* __restrict__ eidx, int E,
                                               const int* __restrict__ mapping,
                                               int* __restrict__ eout) {
  int e = blockIdx.x * 256 + threadIdx.x;
  if (e >= E) return;
  int s = eidx[e], d = eidx[E + e];
  int ms = mapping[s], md = mapping[d];
  bool ok = (ms >= 0) & (md >= 0);
  eout[e]     = ok ? ms : -1;
  eout[E + e] = ok ? md : -1;
}

extern "C" void kernel_launch(void* const* d_in, const int* in_sizes, int n_in,
                              void* d_out, int out_size, void* d_ws, size_t ws_size,
                              hipStream_t stream) {
  const float* x    = (const float*)d_in[0];
  const int*   eidx = (const int*)d_in[1];
  const float* w    = (const float*)d_in[2];

  const int C    = in_sizes[2];          // 256
  const int N    = in_sizes[0] / C;      // 100000
  const int E    = in_sizes[1] / 2;      // 3200000
  const int Ksel = (N + 1) / 2;          // 50000
  const int nb   = (N + 255) / 256;

  char* ws = (char*)d_ws;
  SelState* st   = (SelState*)ws;
  _Float16* hw   = (_Float16*)(ws + 64);        // 32B-aligned for v16h loads
  uint32_t* hist = (uint32_t*)(ws + 1024);
  uint32_t* cntG  = (uint32_t*)(ws + 2048);
  uint32_t* cntE  = (uint32_t*)(ws + 2048 + 4096);
  uint32_t* baseG = (uint32_t*)(ws + 2048 + 8192);
  uint32_t* baseE = (uint32_t*)(ws + 2048 + 12288);
  size_t off = 16384;
  float* scores  = (float*)(ws + off); off += (size_t)N * 4;    off = (off + 255) & ~(size_t)255;
  int*   perm    = (int*)(ws + off);   off += (size_t)Ksel * 4; off = (off + 255) & ~(size_t)255;
  int*   mapping = (int*)(ws + off);

  float* out_x = (float*)d_out;
  int*   out_e = (int*)(out_x + (size_t)Ksel * (size_t)C);

  k_init<<<1, 256, 0, stream>>>(w, C, st, hw, hist, (uint32_t)Ksel);

  const int gScore = (N + 127) / 128;   // 8 waves/block * 16 rows/wave
  k_score<<<gScore, 256, 0, stream>>>(x, hw, st, scores, N);

  for (int shift = 24; shift >= 0; shift -= 8) {
    k_hist<<<nb, 256, 0, stream>>>(scores, N, hist, st, shift);
    k_select<<<1, 256, 0, stream>>>(hist, st, shift);
  }

  k_count<<<nb, 256, 0, stream>>>(scores, N, st, cntG, cntE);
  k_scan<<<1, 1, 0, stream>>>(cntG, cntE, baseG, baseE, nb, st);
  k_scatter<<<nb, 256, 0, stream>>>(scores, N, st, baseG, baseE, perm, mapping);

  k_gather<<<Ksel, 64, 0, stream>>>(x, perm, scores, out_x);
  k_edges<<<(E + 255) / 256, 256, 0, stream>>>(eidx, E, mapping, out_e);
}
